// DGCNN_55130200211664
// MI455X (gfx1250) — compile-verified, matrix operand order checked by reading, not run on hardware
//
#include <hip/hip_runtime.h>

#define DEVINL __device__ __forceinline__

typedef __attribute__((ext_vector_type(16))) __bf16 v16bf;
typedef __attribute__((ext_vector_type(8)))  __bf16 v8bf;
typedef __attribute__((ext_vector_type(8)))  float  v8f;

union V16U { v16bf v; v8bf h[2]; };

static constexpr int BB   = 8;
static constexpr int NN   = 4096;
static constexpr int KNB  = 20;
static constexpr int FDIM = 512;
static constexpr float EPSV = 1e-5f;

DEVINL v8f wmma_bf16(v16bf a, v16bf b, v8f c) {
  // D = A(16xK=32) * B(32x16) + C ; f32 accum
  return __builtin_amdgcn_wmma_f32_16x16x32_bf16(false, a, false, b, (short)0, c, false, false);
}

// 16-bit A/B operand layout (ISA 7.12.2): lane l serves row/col (l%16);
// VGPR0..3 hold K = (l/16)*8 .. +7 ; VGPR4..7 hold K = 16+(l/16)*8 .. +7
// -> two 16B loads of 8 consecutive bf16 from a row-major row.
DEVINL v16bf ld_row16(const __bf16* p, int k0) {
  V16U u;
  u.h[0] = *(const v8bf*)(p + k0);
  u.h[1] = *(const v8bf*)(p + k0 + 16);
  return u.v;
}

template<int CIN>
DEVINL v8bf edge8(const float* ctr, const float* nbr, int c0) {
  v8bf r;
#pragma unroll
  for (int i = 0; i < 8; ++i) {
    int c = c0 + i;
    float v;
    if (c < CIN)            v = nbr[c] - ctr[c];
    else if (c < 2 * CIN)   v = ctr[c - CIN];
    else                    v = 0.f;
    r[i] = (__bf16)v;
  }
  return r;
}

// ---------------------------------------------------------------------------
// KNN: one wave = 32 queries. A tiles resident in regs; stream candidate
// chunks of 16 through WMMA; score = 2*inner - ||xj||^2 (per-query constant
// dropped: ordering-invariant). Top-20 kept sorted in registers per lane.
// ---------------------------------------------------------------------------
template<int CP>
__global__ __launch_bounds__(32)
void knn_kernel(const __bf16* __restrict__ feat, const float* __restrict__ xx,
                int* __restrict__ idx) {
  constexpr int SL = CP / 32;
  const int l = threadIdx.x, ln = l & 15, lh = l >> 4;
  const int b = blockIdx.y;
  const int qbase = blockIdx.x * 32;
  const __bf16* fb = feat + (size_t)b * NN * CP;
  const float*  xb = xx   + (size_t)b * NN;

  v16bf A[2][SL];
#pragma unroll
  for (int qt = 0; qt < 2; ++qt) {
    const __bf16* row = fb + (size_t)(qbase + qt * 16 + ln) * CP;
#pragma unroll
    for (int s = 0; s < SL; ++s) A[qt][s] = ld_row16(row, s * 32 + lh * 8);
  }

  float topv[KNB]; int topi[KNB];
#pragma unroll
  for (int t = 0; t < KNB; ++t) { topv[t] = -3.4e38f; topi[t] = 0; }

  __shared__ float sd[32][17];

  for (int ch = 0; ch < NN / 16; ++ch) {
    const int jb = ch * 16;
    const __bf16* crow = fb + (size_t)(jb + ln) * CP;
    v16bf Bt[SL];
#pragma unroll
    for (int s = 0; s < SL; ++s) Bt[s] = ld_row16(crow, s * 32 + lh * 8);
    const float xxj = xb[jb + ln];
#pragma unroll
    for (int qt = 0; qt < 2; ++qt) {
      v8f acc = {};
#pragma unroll
      for (int s = 0; s < SL; ++s) acc = wmma_bf16(A[qt][s], Bt[s], acc);
      // D layout: lanes 0-15 -> M=v, lanes 16-31 -> M=v+8; N = l%16
#pragma unroll
      for (int v = 0; v < 8; ++v)
        sd[qt * 16 + v + 8 * lh][ln] = 2.0f * acc[v] - xxj;
    }
    // DS ops are in-order per wave: reads below see the writes above.
#pragma unroll
    for (int jj = 0; jj < 16; ++jj) {
      float s = sd[l][jj];
      if (s > topv[KNB - 1]) {
        topv[KNB - 1] = s; topi[KNB - 1] = jb + jj;
#pragma unroll
        for (int t = KNB - 1; t > 0; --t) {
          if (topv[t] > topv[t - 1]) {
            float tv = topv[t]; topv[t] = topv[t - 1]; topv[t - 1] = tv;
            int   ti = topi[t]; topi[t] = topi[t - 1]; topi[t - 1] = ti;
          }
        }
      }
    }
  }
  int* op = idx + ((size_t)b * NN + qbase + l) * KNB;
#pragma unroll
  for (int t = 0; t < KNB; ++t) op[t] = topi[t];
}

// ---------------------------------------------------------------------------
// Edge-conv stage. One wave handles 4 points = 80 edge rows = 5 M-tiles.
// MODE 0: per-channel sum/sumsq of conv1 output (for BN1 batch stats).
// MODE 1: (NCONV==2) recompute conv1 -> BN1+LReLU -> conv2, stats of conv2.
// MODE 2: full pipeline, max over k -> xout (B,N,64) f32.
// ---------------------------------------------------------------------------
template<int CIN, int KP, int NCONV, int MODE>
__global__ __launch_bounds__(32)
void edgeconv_kernel(const float* __restrict__ xf, const int* __restrict__ idx,
                     const __bf16* __restrict__ W1p, const __bf16* __restrict__ W2p,
                     const float* __restrict__ ab1, const float* __restrict__ ab2,
                     float* __restrict__ sums, float* __restrict__ xout) {
  constexpr int SL1 = KP / 32;
  const int l = threadIdx.x, ln = l & 15, lh = l >> 4;
  const int b = blockIdx.y;
  const int pbase = blockIdx.x * 4;
  const float* xb = xf + (size_t)b * NN * CIN;
  const int* ib = idx + ((size_t)b * NN + pbase) * KNB;

  __shared__ float  hbuf[80][65];
  __shared__ __bf16 tb[16][64];

  float a1v[4], c1v[4], a2v[4], c2v[4];
  if (MODE >= 1) {
#pragma unroll
    for (int ot = 0; ot < 4; ++ot) { a1v[ot] = ab1[ot * 16 + ln]; c1v[ot] = ab1[64 + ot * 16 + ln]; }
  }
  if (MODE == 2 && NCONV == 2) {
#pragma unroll
    for (int ot = 0; ot < 4; ++ot) { a2v[ot] = ab2[ot * 16 + ln]; c2v[ot] = ab2[64 + ot * 16 + ln]; }
  }
  float ps1[4] = {0, 0, 0, 0}, ps2[4] = {0, 0, 0, 0};

  for (int t = 0; t < 5; ++t) {
    const int arow = t * 16 + ln;            // row this lane feeds (A operand)
    const int pa = arow / KNB, kk = arow % KNB;
    const int j = ib[pa * KNB + kk];
    const float* ctr = xb + (size_t)(pbase + pa) * CIN;
    const float* nbr = xb + (size_t)j * CIN;

    v16bf A1[SL1];
#pragma unroll
    for (int s = 0; s < SL1; ++s) {
      V16U u;
      u.h[0] = edge8<CIN>(ctr, nbr, s * 32 + lh * 8);
      u.h[1] = edge8<CIN>(ctr, nbr, s * 32 + lh * 8 + 16);
      A1[s] = u.v;
    }
    v8f y1[4];
#pragma unroll
    for (int ot = 0; ot < 4; ++ot) {
      v8f acc = {};
#pragma unroll
      for (int s = 0; s < SL1; ++s)
        acc = wmma_bf16(A1[s], ld_row16(W1p + (size_t)(ot * 16 + ln) * KP, s * 32 + lh * 8), acc);
      y1[ot] = acc;
    }
    if (MODE == 0) {
#pragma unroll
      for (int ot = 0; ot < 4; ++ot)
#pragma unroll
        for (int v = 0; v < 8; ++v) { float y = y1[ot][v]; ps1[ot] += y; ps2[ot] += y * y; }
      continue;
    }
    // BN1 + LReLU(0.2)
#pragma unroll
    for (int ot = 0; ot < 4; ++ot)
#pragma unroll
      for (int v = 0; v < 8; ++v) {
        float h = fmaf(a1v[ot], y1[ot][v], c1v[ot]);
        y1[ot][v] = h >= 0.f ? h : 0.2f * h;
      }
    if (NCONV == 1) {
#pragma unroll
      for (int ot = 0; ot < 4; ++ot)
#pragma unroll
        for (int v = 0; v < 8; ++v)
          hbuf[t * 16 + v + 8 * lh][ot * 16 + ln] = y1[ot][v];
    } else {
      // transpose 16x64 activations through LDS into A layout for conv2
#pragma unroll
      for (int ot = 0; ot < 4; ++ot)
#pragma unroll
        for (int v = 0; v < 8; ++v)
          tb[v + 8 * lh][ot * 16 + ln] = (__bf16)y1[ot][v];
      v16bf A2[2];
#pragma unroll
      for (int s = 0; s < 2; ++s) A2[s] = ld_row16(&tb[ln][0], s * 32 + lh * 8);
      v8f y2[4];
#pragma unroll
      for (int ot = 0; ot < 4; ++ot) {
        v8f acc = {};
#pragma unroll
        for (int s = 0; s < 2; ++s)
          acc = wmma_bf16(A2[s], ld_row16(W2p + (size_t)(ot * 16 + ln) * 64, s * 32 + lh * 8), acc);
        y2[ot] = acc;
      }
      if (MODE == 1) {
#pragma unroll
        for (int ot = 0; ot < 4; ++ot)
#pragma unroll
          for (int v = 0; v < 8; ++v) { float y = y2[ot][v]; ps1[ot] += y; ps2[ot] += y * y; }
      } else {
#pragma unroll
        for (int ot = 0; ot < 4; ++ot)
#pragma unroll
          for (int v = 0; v < 8; ++v) {
            float h = fmaf(a2v[ot], y2[ot][v], c2v[ot]);
            h = h >= 0.f ? h : 0.2f * h;
            hbuf[t * 16 + v + 8 * lh][ot * 16 + ln] = h;
          }
      }
    }
  }

  if (MODE == 0 || (MODE == 1 && NCONV == 2)) {
#pragma unroll
    for (int ot = 0; ot < 4; ++ot) {
      float s1 = ps1[ot] + __shfl_xor(ps1[ot], 16, 32);
      float s2 = ps2[ot] + __shfl_xor(ps2[ot], 16, 32);
      if (lh == 0) {
        atomicAdd(&sums[ot * 16 + ln], s1);
        atomicAdd(&sums[64 + ot * 16 + ln], s2);
      }
    }
    return;
  }
  if (MODE == 2) {
    // max over k: lane owns point l/8, channels (l%8)*8..+7 (disjoint)
    const int pl = l >> 3;
    const int c0 = (l & 7) * 8;
    float* op = xout + ((size_t)b * NN + pbase + pl) * 64;
#pragma unroll
    for (int ci = 0; ci < 8; ++ci) {
      const int c = c0 + ci;
      float m = -3.4e38f;
#pragma unroll
      for (int kk = 0; kk < KNB; ++kk) {
        float v = hbuf[pl * KNB + kk][c];
        m = v > m ? v : m;
      }
      op[c] = m;
    }
  }
}

// ---------------------------------------------------------------------------
// conv6: (B*N) x 192 -> 512, raw output into d_out local region + BN stats.
// ---------------------------------------------------------------------------
__global__ __launch_bounds__(256)
void conv6_kernel(const __bf16* __restrict__ catb, const __bf16* __restrict__ W6p,
                  float* __restrict__ outLocal, float* __restrict__ sums) {
  __shared__ float lstat[1024];
  const int tid = threadIdx.x;
  for (int i = tid; i < 1024; i += 256) lstat[i] = 0.f;
  __syncthreads();
  const int l = tid & 31, ln = l & 15, lh = (l >> 4) & 1;
  const int w = tid >> 5;
  const int b = blockIdx.y;
  const int mbase = blockIdx.x * 128 + w * 16;

  v16bf A[6];
  const __bf16* row = catb + ((size_t)b * NN + mbase + ln) * 192;
#pragma unroll
  for (int s = 0; s < 6; ++s) A[s] = ld_row16(row, s * 32 + lh * 8);

  for (int ot = 0; ot < 32; ++ot) {
    v8f acc = {};
#pragma unroll
    for (int s = 0; s < 6; ++s)
      acc = wmma_bf16(A[s], ld_row16(W6p + (size_t)(ot * 16 + ln) * 192, s * 32 + lh * 8), acc);
    const int fd = ot * 16 + ln;
    float s1 = 0.f, s2 = 0.f;
#pragma unroll
    for (int v = 0; v < 8; ++v) {
      const int pr = mbase + v + 8 * lh;
      outLocal[((size_t)b * NN + pr) * FDIM + fd] = acc[v];
      s1 += acc[v]; s2 += acc[v] * acc[v];
    }
    atomicAdd(&lstat[fd], s1);
    atomicAdd(&lstat[512 + fd], s2);
  }
  __syncthreads();
  for (int i = tid; i < 1024; i += 256) atomicAdd(&sums[i], lstat[i]);
}

__global__ void bn6_kernel(float* __restrict__ outLocal, const float* __restrict__ ab6) {
  size_t i = (size_t)blockIdx.x * blockDim.x + threadIdx.x;
  if (i >= (size_t)BB * NN * FDIM) return;
  int fd = (int)(i & (FDIM - 1));
  float h = fmaf(ab6[fd], outLocal[i], ab6[FDIM + fd]);
  outLocal[i] = h >= 0.f ? h : 0.2f * h;
}

__global__ __launch_bounds__(256)
void gx_kernel(const float* __restrict__ outLocal, float* __restrict__ gx) {
  const int b = blockIdx.y;
  const int fd = blockIdx.x * 64 + (threadIdx.x & 63);
  const int g = threadIdx.x >> 6;
  float m = -3.4e38f;
  for (int n = g; n < NN; n += 4) {
    float v = outLocal[((size_t)b * NN + n) * FDIM + fd];
    m = v > m ? v : m;
  }
  __shared__ float red[256];
  red[threadIdx.x] = m;
  __syncthreads();
  if (g == 0) {
#pragma unroll
    for (int q = 1; q < 4; ++q) { float v = red[threadIdx.x + 64 * q]; m = v > m ? v : m; }
    gx[b * FDIM + fd] = m;
  }
}

__global__ __launch_bounds__(512)
void head_kernel(const float* __restrict__ gx,
                 const float* __restrict__ LW1, const float* __restrict__ Lb1,
                 const float* __restrict__ LW2, const float* __restrict__ Lb2,
                 const float* __restrict__ gg, const float* __restrict__ gb,
                 float* __restrict__ gout) {
  __shared__ float hs[BB * FDIM];
  const int f = threadIdx.x;
  for (int b = 0; b < BB; ++b) {
    float acc = Lb1[f];
    const float* wrow = LW1 + (size_t)f * FDIM;
    const float* gr = gx + b * FDIM;
    for (int c = 0; c < FDIM; ++c) acc = fmaf(gr[c], wrow[c], acc);
    hs[b * FDIM + f] = acc;
  }
  __syncthreads();
  {
    float s = 0.f, ss = 0.f;
#pragma unroll
    for (int b = 0; b < BB; ++b) { float v = hs[b * FDIM + f]; s += v; ss += v * v; }
    float mean = s * (1.f / BB);
    float var = ss * (1.f / BB) - mean * mean;
    float a = gg[f] * rsqrtf(var + EPSV);
    float c = gb[f] - a * mean;
#pragma unroll
    for (int b = 0; b < BB; ++b) {
      float v = fmaf(a, hs[b * FDIM + f], c);
      hs[b * FDIM + f] = v >= 0.f ? v : 0.01f * v;
    }
  }
  __syncthreads();
  for (int b = 0; b < BB; ++b) {
    float acc = Lb2[f];
    const float* wrow = LW2 + (size_t)f * FDIM;
    const float* hr = hs + b * FDIM;
    for (int c = 0; c < FDIM; ++c) acc = fmaf(hr[c], wrow[c], acc);
    gout[b * FDIM + f] = acc;
  }
}

// ------------------------- small utility kernels ---------------------------
__global__ void pack_weight_kernel(const float* __restrict__ W, __bf16* __restrict__ Wp,
                                   int O, int C, int Cp) {
  int i = blockIdx.x * blockDim.x + threadIdx.x;
  if (i >= O * Cp) return;
  int o = i / Cp, c = i % Cp;
  Wp[i] = (__bf16)(c < C ? W[o * C + c] : 0.f);
}

__global__ void pack_x_kernel(const float* __restrict__ x, __bf16* __restrict__ fb,
                              float* __restrict__ xx) {
  int i = blockIdx.x * blockDim.x + threadIdx.x;
  if (i >= BB * NN) return;
  float a = x[i * 3 + 0], b = x[i * 3 + 1], c = x[i * 3 + 2];
  __bf16* o = fb + (size_t)i * 32;
  o[0] = (__bf16)a; o[1] = (__bf16)b; o[2] = (__bf16)c;
  for (int k = 3; k < 32; ++k) o[k] = (__bf16)0.f;
  xx[i] = a * a + b * b + c * c;
}

__global__ void pack_feat64_kernel(const float* __restrict__ xf, __bf16* __restrict__ knnb,
                                   float* __restrict__ xx, __bf16* __restrict__ catb, int cofs) {
  int i = blockIdx.x * blockDim.x + threadIdx.x;
  if (i >= BB * NN) return;
  const float* r = xf + (size_t)i * 64;
  __bf16* cb = catb + (size_t)i * 192 + cofs;
  float s = 0.f;
  for (int c = 0; c < 64; ++c) {
    float v = r[c]; s += v * v;
    __bf16 bv = (__bf16)v;
    if (knnb) knnb[(size_t)i * 64 + c] = bv;
    cb[c] = bv;
  }
  if (xx) xx[i] = s;
}

__global__ void zero_kernel(float* p, int n) {
  int i = blockIdx.x * blockDim.x + threadIdx.x;
  if (i < n) p[i] = 0.f;
}

__global__ void finalize_bn_kernel(const float* __restrict__ sums, const float* __restrict__ g,
                                   const float* __restrict__ bb, float* __restrict__ ab,
                                   int C, float invcnt) {
  int o = blockIdx.x * blockDim.x + threadIdx.x;
  if (o >= C) return;
  float mean = sums[o] * invcnt;
  float var = sums[C + o] * invcnt - mean * mean;
  float a = g[o] * rsqrtf(var + EPSV);
  ab[o] = a;
  ab[C + o] = bb[o] - a * mean;
}

// ---------------------------------------------------------------------------
extern "C" void kernel_launch(void* const* d_in, const int* in_sizes, int n_in,
                              void* d_out, int out_size, void* d_ws, size_t ws_size,
                              hipStream_t stream) {
  (void)in_sizes; (void)n_in; (void)out_size; (void)ws_size;
  const float* x   = (const float*)d_in[0];
  const float* W1  = (const float*)d_in[1];
  const float* W2  = (const float*)d_in[2];
  const float* W3  = (const float*)d_in[3];
  const float* W4  = (const float*)d_in[4];
  const float* W5  = (const float*)d_in[5];
  const float* W6  = (const float*)d_in[6];
  const float* LW1 = (const float*)d_in[7];
  const float* Lb1 = (const float*)d_in[8];
  const float* LW2 = (const float*)d_in[9];
  const float* Lb2 = (const float*)d_in[10];
  const float* gg  = (const float*)d_in[11];
  const float* gb  = (const float*)d_in[12];
  const float* gA[6] = {(const float*)d_in[13], (const float*)d_in[15], (const float*)d_in[17],
                        (const float*)d_in[19], (const float*)d_in[21], (const float*)d_in[23]};
  const float* bA[6] = {(const float*)d_in[14], (const float*)d_in[16], (const float*)d_in[18],
                        (const float*)d_in[20], (const float*)d_in[22], (const float*)d_in[24]};
  float* out = (float*)d_out;
  float* outLocal = out + BB * FDIM;

  char* base = (char*)d_ws;
  size_t off = 0;
  auto alloc = [&](size_t bytes) -> void* {
    void* p = base + off;
    off = (off + bytes + 255) & ~(size_t)255;
    return p;
  };
  __bf16* W1p = (__bf16*)alloc(64 * 32 * 2);
  __bf16* W2p = (__bf16*)alloc(64 * 64 * 2);
  __bf16* W3p = (__bf16*)alloc(64 * 128 * 2);
  __bf16* W4p = (__bf16*)alloc(64 * 64 * 2);
  __bf16* W5p = (__bf16*)alloc(64 * 128 * 2);
  __bf16* W6p = (__bf16*)alloc((size_t)512 * 192 * 2);
  __bf16* knnb1  = (__bf16*)alloc((size_t)BB * NN * 32 * 2);
  __bf16* knnb64 = (__bf16*)alloc((size_t)BB * NN * 64 * 2);
  float*  xxb  = (float*)alloc((size_t)BB * NN * 4);
  int*    idxb = (int*)alloc((size_t)BB * NN * KNB * 4);
  float*  x1f  = (float*)alloc((size_t)BB * NN * 64 * 4);
  float*  x2f  = (float*)alloc((size_t)BB * NN * 64 * 4);
  float*  x3f  = (float*)alloc((size_t)BB * NN * 64 * 4);
  __bf16* catb = (__bf16*)alloc((size_t)BB * NN * 192 * 2);
  float*  sums = (float*)alloc(1024 * 4);
  float*  ab1  = (float*)alloc(128 * 4);
  float*  ab2  = (float*)alloc(128 * 4);
  float*  ab6  = (float*)alloc(1024 * 4);
  float*  gxb  = (float*)alloc((size_t)BB * FDIM * 4);

  const float invE = 1.f / (float)((size_t)BB * NN * KNB);  // edge-conv BN count
  const float invP = 1.f / (float)((size_t)BB * NN);        // conv6 BN count
  const dim3 egrid(NN / 4, BB), kgrid(NN / 32, BB);
  const int PT = (BB * NN + 255) / 256;

  // weight packing (bf16, K zero-padded to multiples of 32)
  pack_weight_kernel<<<(64 * 32 + 255) / 256, 256, 0, stream>>>(W1, W1p, 64, 6, 32);
  pack_weight_kernel<<<(64 * 64 + 255) / 256, 256, 0, stream>>>(W2, W2p, 64, 64, 64);
  pack_weight_kernel<<<(64 * 128 + 255) / 256, 256, 0, stream>>>(W3, W3p, 64, 128, 128);
  pack_weight_kernel<<<(64 * 64 + 255) / 256, 256, 0, stream>>>(W4, W4p, 64, 64, 64);
  pack_weight_kernel<<<(64 * 128 + 255) / 256, 256, 0, stream>>>(W5, W5p, 64, 128, 128);
  pack_weight_kernel<<<(512 * 192 + 255) / 256, 256, 0, stream>>>(W6, W6p, 512, 192, 192);

  // ---- stage 1 (CIN=3, edge K padded to 32) ----
  pack_x_kernel<<<PT, 256, 0, stream>>>(x, knnb1, xxb);
  knn_kernel<32><<<kgrid, 32, 0, stream>>>(knnb1, xxb, idxb);
  zero_kernel<<<1, 128, 0, stream>>>(sums, 128);
  edgeconv_kernel<3, 32, 2, 0><<<egrid, 32, 0, stream>>>(x, idxb, W1p, W2p, ab1, ab2, sums, x1f);
  finalize_bn_kernel<<<1, 64, 0, stream>>>(sums, gA[0], bA[0], ab1, 64, invE);
  zero_kernel<<<1, 128, 0, stream>>>(sums, 128);
  edgeconv_kernel<3, 32, 2, 1><<<egrid, 32, 0, stream>>>(x, idxb, W1p, W2p, ab1, ab2, sums, x1f);
  finalize_bn_kernel<<<1, 64, 0, stream>>>(sums, gA[1], bA[1], ab2, 64, invE);
  edgeconv_kernel<3, 32, 2, 2><<<egrid, 32, 0, stream>>>(x, idxb, W1p, W2p, ab1, ab2, sums, x1f);
  pack_feat64_kernel<<<PT, 256, 0, stream>>>(x1f, knnb64, xxb, catb, 0);

  // ---- stage 2 (CIN=64, edge K=128) ----
  knn_kernel<64><<<kgrid, 32, 0, stream>>>(knnb64, xxb, idxb);
  zero_kernel<<<1, 128, 0, stream>>>(sums, 128);
  edgeconv_kernel<64, 128, 2, 0><<<egrid, 32, 0, stream>>>(x1f, idxb, W3p, W4p, ab1, ab2, sums, x2f);
  finalize_bn_kernel<<<1, 64, 0, stream>>>(sums, gA[2], bA[2], ab1, 64, invE);
  zero_kernel<<<1, 128, 0, stream>>>(sums, 128);
  edgeconv_kernel<64, 128, 2, 1><<<egrid, 32, 0, stream>>>(x1f, idxb, W3p, W4p, ab1, ab2, sums, x2f);
  finalize_bn_kernel<<<1, 64, 0, stream>>>(sums, gA[3], bA[3], ab2, 64, invE);
  edgeconv_kernel<64, 128, 2, 2><<<egrid, 32, 0, stream>>>(x1f, idxb, W3p, W4p, ab1, ab2, sums, x2f);
  pack_feat64_kernel<<<PT, 256, 0, stream>>>(x2f, knnb64, xxb, catb, 64);

  // ---- stage 3 (CIN=64, single conv) ----
  knn_kernel<64><<<kgrid, 32, 0, stream>>>(knnb64, xxb, idxb);
  zero_kernel<<<1, 128, 0, stream>>>(sums, 128);
  edgeconv_kernel<64, 128, 1, 0><<<egrid, 32, 0, stream>>>(x2f, idxb, W5p, W4p, ab1, ab2, sums, x3f);
  finalize_bn_kernel<<<1, 64, 0, stream>>>(sums, gA[4], bA[4], ab1, 64, invE);
  edgeconv_kernel<64, 128, 1, 2><<<egrid, 32, 0, stream>>>(x2f, idxb, W5p, W4p, ab1, ab2, sums, x3f);
  pack_feat64_kernel<<<PT, 256, 0, stream>>>(x3f, nullptr, nullptr, catb, 128);

  // ---- conv6 + BN6 + gx ----
  zero_kernel<<<4, 256, 0, stream>>>(sums, 1024);
  conv6_kernel<<<dim3(NN / 128, BB), 256, 0, stream>>>(catb, W6p, outLocal, sums);
  finalize_bn_kernel<<<2, 256, 0, stream>>>(sums, gA[5], bA[5], ab6, 512, invP);
  {
    size_t total = (size_t)BB * NN * FDIM;
    bn6_kernel<<<(unsigned)((total + 255) / 256), 256, 0, stream>>>(outLocal, ab6);
  }
  gx_kernel<<<dim3(FDIM / 64, BB), 256, 0, stream>>>(outLocal, gxb);

  // ---- FC head ----
  head_kernel<<<1, 512, 0, stream>>>(gxb, LW1, Lb1, LW2, Lb2, gg, gb, out);
}
